// diagonal_LSTM_63333587747410
// MI455X (gfx1250) — matrix operand order repaired and checked by reading, not compiled, not run on hardware
//
#include <hip/hip_runtime.h>

typedef __bf16 bf16_t;
typedef __attribute__((ext_vector_type(16))) __bf16 v16bf;
typedef __attribute__((ext_vector_type(8)))  __bf16 v8bf;
typedef __attribute__((ext_vector_type(8)))  float  v8f;

#define C_DIM 128
#define H_DIM 64
#define W_DIM 64
#define WS    127            // W + H - 1 skewed width
#define KT    12             // 384 / 32 k-tiles
#define HH_STRIDE 392        // 384 + 8 pad (bf16 elems), conflict-free b128 frag loads
#define G_STRIDE  72         // 64 + 8 pad (bf16 elems), 144B rows -> 16B aligned vec reads
#define C_STRIDE  68         // 64 + 4 pad (f32 elems), 272B rows -> 16B aligned vec reads
#define NTHREADS 512         // 16 waves (wave32)

__device__ __forceinline__ float fast_sigmoid(float x) {
  float e = __builtin_amdgcn_exp2f(-1.4426950408889634f * x);
  return __builtin_amdgcn_rcpf(1.0f + e);
}
__device__ __forceinline__ float fast_tanh(float x) {
  float e = __builtin_amdgcn_exp2f(2.8853900817779268f * x);
  return 1.0f - 2.0f * __builtin_amdgcn_rcpf(e + 1.0f);
}
__device__ __forceinline__ int clamp_w(int w) {
  return (w < 0) ? 0 : ((w > W_DIM - 1) ? (W_DIM - 1) : w);
}

// Concatenated weight matrix A[512][384]: k<128 -> w_ss[:, :, 0] (h_shift),
// 128..255 -> w_ss[:, :, 1] (h), 256..383 -> w_is (x slice)
__device__ __forceinline__ float wload(const float* __restrict__ w_is,
                                       const float* __restrict__ w_ss,
                                       int row, int k) {
  if (k < 128)      return w_ss[(row * 128 + k) * 2 + 0];
  else if (k < 256) return w_ss[(row * 128 + (k - 128)) * 2 + 1];
  else              return w_is[row * 128 + (k - 256)];
}

__global__ __launch_bounds__(NTHREADS, 1)
void diag_lstm_kernel(const float* __restrict__ x,
                      const float* __restrict__ w_is,
                      const float* __restrict__ b_is,
                      const float* __restrict__ w_ss,
                      const float* __restrict__ b_ss,
                      float* __restrict__ out) {
  // B-operand buffer, K-major per column h: rows [0,128)=h_shift, [128,256)=h, [256,384)=x_t
  __shared__ __align__(16) bf16_t hh[H_DIM][HH_STRIDE];      // ~50 KB
  __shared__ __align__(16) bf16_t gact[4 * C_DIM][G_STRIDE]; // raw gates (bf16), ~72 KB
  __shared__ __align__(16) float  cmem[C_DIM][C_STRIDE];     // cell state, ~34 KB

  const int tid  = threadIdx.x;
  const int b    = blockIdx.x;
  const int lane = tid & 31;
  const int wave = tid >> 5;

  // ---- zero hh and cmem (h/c state starts at 0; hh col 0 h_shift rows stay 0) ----
  {
    unsigned int* p = (unsigned int*)&hh[0][0];
    const int n = (H_DIM * HH_STRIDE) / 2;
    for (int i = tid; i < n; i += NTHREADS) p[i] = 0u;
    unsigned int* q = (unsigned int*)&cmem[0][0];
    const int m = C_DIM * C_STRIDE;
    for (int i = tid; i < m; i += NTHREADS) q[i] = 0u;
  }

  // ---- persistent A fragments in VGPRs: wave owns M-tiles {2*wave, 2*wave+1} ----
  const int m0   = wave * 2;
  const int rowA = lane & 15;
  const int kbA  = (lane < 16) ? 0 : 8;     // 16-bit A 16x32 layout
  v16bf afrag[2][KT];
  #pragma unroll
  for (int mi = 0; mi < 2; ++mi) {
    const int row = (m0 + mi) * 16 + rowA;
    #pragma unroll
    for (int kt = 0; kt < KT; ++kt) {
      union { v16bf v; bf16_t a[16]; } u;
      #pragma unroll
      for (int e = 0; e < 16; ++e) {
        // V0..3: K = kb..kb+7 ; V4..7: K = kb+16..kb+23
        const int k = kt * 32 + kbA + ((e < 8) ? e : (8 + e));
        u.a[e] = (bf16_t)wload(w_is, w_ss, row, k);
      }
      afrag[mi][kt] = u.v;
    }
  }

  // ---- pointwise ownership: thread -> (channel, 16 h values) ----
  const int ch    = tid >> 2;
  const int hbase = (tid & 3) * 16;

  // per-thread gate biases (folded into pointwise activation; only 4 VGPRs)
  const float bo = b_is[ch]             + b_ss[ch];
  const float bf = b_is[C_DIM + ch]     + b_ss[C_DIM + ch];
  const float bi = b_is[2 * C_DIM + ch] + b_ss[2 * C_DIM + ch];
  const float bg = b_is[3 * C_DIM + ch] + b_ss[3 * C_DIM + ch];

  const float* xb = x   + ((size_t)b * C_DIM + ch) * (H_DIM * W_DIM);
  float*       ob = out + ((size_t)b * C_DIM + ch) * (H_DIM * W_DIM);
  float*       crow = &cmem[ch][hbase];

  __syncthreads();  // hh/cmem fully zeroed before x staging overwrites hh rows 256..383

  // ---- stage x(t=0): branchless clamped load + select ----
  #pragma unroll
  for (int j = 0; j < 16; ++j) {
    const int h = hbase + j;
    const int w = 0 - h;
    float v = xb[h * W_DIM + clamp_w(w)];
    v = (w >= 0 && w < W_DIM) ? v : 0.0f;
    hh[h][2 * C_DIM + ch] = (bf16_t)v;
  }
  __syncthreads();

  const int colB = lane & 15;
  const int kbB  = (lane < 16) ? 0 : 8;     // 16-bit B 32x16 layout

  for (int t = 0; t < WS; ++t) {
    // ---- register-free prefetch of next x slice; GEMM below hides the latency ----
    if (t + 1 < WS) {
      #pragma unroll
      for (int j = 0; j < 16; ++j) {
        const int h = hbase + j;
        __builtin_prefetch(&xb[h * W_DIM + clamp_w(t + 1 - h)], 0, 3);
      }
    }

    // ---- per-step GEMM: gates[512][64] = A[512x384] x hh[384x64] (bias added later) ----
    #pragma unroll
    for (int nt = 0; nt < 4; ++nt) {
      v8f acc0 = {0.f, 0.f, 0.f, 0.f, 0.f, 0.f, 0.f, 0.f};
      v8f acc1 = {0.f, 0.f, 0.f, 0.f, 0.f, 0.f, 0.f, 0.f};
      const bf16_t* bp = &hh[nt * 16 + colB][kbB];
      #pragma unroll
      for (int kt = 0; kt < KT; ++kt) {
        const v8bf lo = *(const v8bf*)(bp + kt * 32);
        const v8bf hi = *(const v8bf*)(bp + kt * 32 + 16);
        const v16bf bfrag = __builtin_shufflevector(lo, hi,
            0, 1, 2, 3, 4, 5, 6, 7, 8, 9, 10, 11, 12, 13, 14, 15);
        acc0 = __builtin_amdgcn_wmma_f32_16x16x32_bf16(
            false, afrag[0][kt], false, bfrag, (short)0, acc0, false, false);
        acc1 = __builtin_amdgcn_wmma_f32_16x16x32_bf16(
            false, afrag[1][kt], false, bfrag, (short)0, acc1, false, false);
      }
      // store raw gates (bf16) in fragment layout; activation happens pointwise
      const int row0 = m0 * 16 + ((lane < 16) ? 0 : 8);
      const int colo = nt * 16 + colB;
      #pragma unroll
      for (int v = 0; v < 8; ++v) {
        gact[row0 + v][colo]      = (bf16_t)acc0[v];
        gact[row0 + v + 16][colo] = (bf16_t)acc1[v];
      }
    }
    __syncthreads();

    // ---- pointwise LSTM update + stage next x slice (loads hit prefetched cache) ----
    {
      // issue next-x loads first; independent of gact reads, scheduler overlaps them
      if (t + 1 < WS) {
        #pragma unroll
        for (int j = 0; j < 16; ++j) {
          const int h = hbase + j;
          const int w = t + 1 - h;
          float v = xb[h * W_DIM + clamp_w(w)];
          v = (w >= 0 && w < W_DIM) ? v : 0.0f;
          hh[h][2 * C_DIM + ch] = (bf16_t)v;      // x region for next step
        }
      }

      const bf16_t* po = &gact[ch][hbase];
      const bf16_t* pf = &gact[C_DIM + ch][hbase];
      const bf16_t* pi = &gact[2 * C_DIM + ch][hbase];
      const bf16_t* pg = &gact[3 * C_DIM + ch][hbase];

      // two halves of 8: halves the live v8bf gate vectors; c state lives in LDS
      #pragma unroll
      for (int half = 0; half < 2; ++half) {
        const int jb = half * 8;
        const v8bf ov = *(const v8bf*)(po + jb);
        const v8bf fv = *(const v8bf*)(pf + jb);
        const v8bf iv = *(const v8bf*)(pi + jb);
        const v8bf gv = *(const v8bf*)(pg + jb);
        union { float4 v4[2]; float f[8]; } cu;
        cu.v4[0] = *(const float4*)(crow + jb);
        cu.v4[1] = *(const float4*)(crow + jb + 4);
        #pragma unroll
        for (int q = 0; q < 8; ++q) {
          const float og = fast_sigmoid((float)ov[q] + bo);
          const float fg = fast_sigmoid((float)fv[q] + bf);
          const float ii = fast_sigmoid((float)iv[q] + bi);
          const float gg = fast_tanh   ((float)gv[q] + bg);
          const float cn = fg * cu.f[q] + ii * gg;
          cu.f[q] = cn;
          const float  hn = og * fast_tanh(cn);
          const bf16_t hb = (bf16_t)hn;
          const int h = hbase + jb + q;
          hh[h][C_DIM + ch] = hb;                      // h region (W1 term)
          if (h + 1 < H_DIM) hh[h + 1][ch] = hb;       // h_shift region (W0 term)
          const int w = t - h;                          // unskew on the fly
          if (w >= 0 && w < W_DIM) ob[h * W_DIM + w] = hn;
        }
        *(float4*)(crow + jb)     = cu.v4[0];
        *(float4*)(crow + jb + 4) = cu.v4[1];
      }
    }
    __syncthreads();
  }
}

extern "C" void kernel_launch(void* const* d_in, const int* in_sizes, int n_in,
                              void* d_out, int out_size, void* d_ws, size_t ws_size,
                              hipStream_t stream) {
  const float* x    = (const float*)d_in[0];
  const float* w_is = (const float*)d_in[1];
  const float* b_is = (const float*)d_in[2];
  const float* w_ss = (const float*)d_in[3];
  const float* b_ss = (const float*)d_in[4];
  float* out = (float*)d_out;

  const int batches = in_sizes[0] / (C_DIM * H_DIM * W_DIM);  // 16
  dim3 grid(batches), block(NTHREADS);
  hipLaunchKernelGGL(diag_lstm_kernel, grid, block, 0, stream,
                     x, w_is, b_is, w_ss, b_ss, out);
}
